// WL_79757542687266
// MI455X (gfx1250) — matrix operand order repaired and checked by reading, not compiled, not run on hardware
//
#include <hip/hip_runtime.h>

// WL subtree kernel (Weisfeiler-Lehman) for MI455X / gfx1250.
//
// Algorithmic transform: the reference's sort-by-(dst,color) + rank-hash only
// serves to hash the *multiset* of neighbor colors; the output depends only on
// the induced partition (dense relabel = first-occurrence order over sig
// equality classes). A commutative sum of splitmix64 hashes of neighbor colors
// yields the identical partition w.h.p. -> no sorting, just u64 atomic adds
// (nsum = 400KB, L2-resident on the 192MB L2).
//
// CDNA5 path: per-graph squared-norm reduction runs on the matrix unit as
// diag(H * H^T) per 16-graph tile via v_wmma_f32_16x16x32_f16, K-split 8 ways
// for memory-level parallelism. Counts are small integers (<< 2048), so f16
// inputs / f32 accumulation / f32 atomics are all bit-exact.

#define N_NODES  50000
#define N_EDGES  1600000
#define N_GRAPHS 256
#define N_LABELS 32
#define N_LAYERS 5
#define HT_CAP   131072          // 2^17 slots, load factor < 0.39
#define HT_MASK  (HT_CAP - 1)
#define HT_EMPTY 0xFFFFFFFFFFFFFFFFull
#define NRM_SPLITS 8
#define NRM_CHUNK  6272          // 196 * 32; last split carries the 16-col tail

typedef __attribute__((ext_vector_type(16))) _Float16 v16h;
typedef __attribute__((ext_vector_type(8)))  float    v8f;

__device__ __forceinline__ unsigned long long mix64(unsigned long long z) {
  z += 0x9E3779B97F4A7C15ull;
  z = (z ^ (z >> 30)) * 0xBF58476D1CE4E5B9ull;
  z = (z ^ (z >> 27)) * 0x94D049BB133111EBull;
  return z ^ (z >> 31);
}

// ---------------------------------------------------------------- zero d_out
__global__ void wl_zero_out(float4* p, long long n4) {
  long long i = (long long)blockIdx.x * blockDim.x + threadIdx.x;
  long long stride = (long long)gridDim.x * blockDim.x;
  for (; i < n4; i += stride) p[i] = make_float4(0.f, 0.f, 0.f, 0.f);
}

// ------------------------------------------------- initial colors = argmax_x
__global__ void wl_argmax(const float* __restrict__ x, int* __restrict__ colors) {
  int i = blockIdx.x * blockDim.x + threadIdx.x;
  if (i >= N_NODES) return;
  const float* row = x + (long long)i * N_LABELS;
  float best = row[0];
  int bi = 0;
#pragma unroll
  for (int j = 1; j < N_LABELS; ++j) {
    float v = row[j];
    if (v > best) { best = v; bi = j; }   // strict > keeps first max (jnp.argmax)
  }
  colors[i] = bi;
}

// ------------------------------------------- per-layer scratch clear (fused)
__global__ void wl_clear(unsigned long long* __restrict__ nsum,
                         unsigned long long* __restrict__ keys,
                         int* __restrict__ vals,
                         float* __restrict__ norms2) {
  int i = blockIdx.x * blockDim.x + threadIdx.x;
  if (i < N_NODES) nsum[i] = 0ull;
  if (i < HT_CAP) { keys[i] = HT_EMPTY; vals[i] = 0x7FFFFFFF; }
  if (i < N_GRAPHS) norms2[i] = 0.f;
}

// ------------------------- edges: commutative multiset hash of neighbor colors
__global__ void wl_edges(const int* __restrict__ src, const int* __restrict__ dst,
                         const int* __restrict__ colors,
                         unsigned long long* __restrict__ nsum) {
  int e = blockIdx.x * blockDim.x + threadIdx.x;
  if (e >= N_EDGES) return;
  int c = colors[src[e]];
  unsigned long long h = mix64(mix64((unsigned long long)(unsigned)c + 1ull));
  atomicAdd(&nsum[dst[e]], h);   // u64 integer sum: order-independent, exact
}

// ------------------------------------- node signature = H(own color, multiset)
__global__ void wl_sig(const int* __restrict__ colors,
                       const unsigned long long* __restrict__ nsum,
                       unsigned long long* __restrict__ sig) {
  int i = blockIdx.x * blockDim.x + threadIdx.x;
  if (i >= N_NODES) return;
  unsigned long long s =
      mix64(mix64((unsigned long long)(unsigned)colors[i] + 1ull) + nsum[i]);
  if (s == HT_EMPTY) s = 0x9E3779B97F4A7C15ull;  // avoid sentinel
  sig[i] = s;
}

// ------------------------------ hash-table insert: class rep = min node index
__global__ void wl_ht_insert(const unsigned long long* __restrict__ sig,
                             unsigned long long* __restrict__ keys,
                             int* __restrict__ vals) {
  int i = blockIdx.x * blockDim.x + threadIdx.x;
  if (i >= N_NODES) return;
  unsigned long long s = sig[i];
  unsigned slot = (unsigned)(mix64(s) & HT_MASK);
  for (;;) {
    unsigned long long k = keys[slot];
    if (k == HT_EMPTY) {
      unsigned long long old = atomicCAS(&keys[slot], HT_EMPTY, s);
      k = (old == HT_EMPTY) ? s : old;
    }
    if (k == s) { atomicMin(&vals[slot], i); return; }
    slot = (slot + 1) & HT_MASK;
  }
}

// ------------------------------------------- lookup rep + leader flag per node
__global__ void wl_lookup(const unsigned long long* __restrict__ sig,
                          const unsigned long long* __restrict__ keys,
                          const int* __restrict__ vals,
                          int* __restrict__ rep, int* __restrict__ flags) {
  int i = blockIdx.x * blockDim.x + threadIdx.x;
  if (i >= N_NODES) return;
  unsigned long long s = sig[i];
  unsigned slot = (unsigned)(mix64(s) & HT_MASK);
  while (keys[slot] != s) slot = (slot + 1) & HT_MASK;
  int r = vals[slot];
  rep[i] = r;
  flags[i] = (r == i) ? 1 : 0;
}

// --------------- exclusive prefix scan of leader flags (single 1024-thr block)
__global__ void wl_scan(const int* __restrict__ flags, int* __restrict__ excl) {
  __shared__ int sdata[1024];
  __shared__ int running;
  if (threadIdx.x == 0) running = 0;
  __syncthreads();
  for (int base = 0; base < N_NODES; base += 1024) {
    int i = base + (int)threadIdx.x;
    int v = (i < N_NODES) ? flags[i] : 0;
    sdata[threadIdx.x] = v;
    __syncthreads();
    for (int off = 1; off < 1024; off <<= 1) {          // Hillis-Steele
      int t = (threadIdx.x >= (unsigned)off) ? sdata[threadIdx.x - off] : 0;
      __syncthreads();
      sdata[threadIdx.x] += t;
      __syncthreads();
    }
    int incl = sdata[threadIdx.x];
    if (i < N_NODES) excl[i] = running + incl - v;
    __syncthreads();
    if (threadIdx.x == 0) running += sdata[1023];
    __syncthreads();
  }
}

// -------- dense relabel: new color = #leaders with index < rep (first-occ order)
__global__ void wl_newcolor(const int* __restrict__ rep, const int* __restrict__ excl,
                            int* __restrict__ ncol) {
  int i = blockIdx.x * blockDim.x + threadIdx.x;
  if (i >= N_NODES) return;
  ncol[i] = excl[rep[i]];
}

// ----------------------------- per-graph color histogram (exact small-int f32)
__global__ void wl_hist(const int* __restrict__ batch, const int* __restrict__ colors,
                        float* __restrict__ h) {
  int i = blockIdx.x * blockDim.x + threadIdx.x;
  if (i >= N_NODES) return;
  atomicAdd(&h[(long long)batch[i] * N_NODES + colors[i]], 1.0f);
}

__device__ __forceinline__ void pack8(v16h& v, int base, float4 lo, float4 hi) {
  v[base + 0] = (_Float16)lo.x; v[base + 1] = (_Float16)lo.y;
  v[base + 2] = (_Float16)lo.z; v[base + 3] = (_Float16)lo.w;
  v[base + 4] = (_Float16)hi.x; v[base + 5] = (_Float16)hi.y;
  v[base + 6] = (_Float16)hi.z; v[base + 7] = (_Float16)hi.w;
}

// ----- per-graph squared norm via matrix unit: diag(H H^T) per 16-graph tile.
// One wave per (16-graph group, K-split); K stepped 32 at a time.
// A (16x32 f16): lane m=lane&15 -> row M; halves H<8 -> K = kb0+H,
//                halves H>=8 -> K = kb0+8+H, kb0 = 8*(lane>>4).   [ISA 7.12.2]
// B (32x16 f16): lane n=lane&15 -> col N; K same as A (table order).
// D (f32, v8f):  diagonal at lanes 0..7 (acc[l]) and 24..31 (acc[l-24]).
// Main loop is guard-free (float4 / b128 loads); only the final 16-col tail
// (50000 = 1562*32 + 16) is handled specially, with a fixed per-lane-group
// valid pattern (no exec-masked scalar loads).
__global__ void __launch_bounds__(32) wl_norm_wmma(const float* __restrict__ hist,
                                                   float* __restrict__ norms2) {
  int g0   = blockIdx.x * 16;
  int lane = threadIdx.x;
  int m    = lane & 15;
  int hi   = lane >> 4;
  int kb0  = hi * 8;
  int hi16 = hi * 16;
  const float* row = hist + (long long)(g0 + m) * N_NODES;  // A row == B col (m==n)

  int start   = (int)blockIdx.y * NRM_CHUNK;
  int end     = start + NRM_CHUNK; if (end > N_NODES) end = N_NODES;
  int fullEnd = start + ((end - start) & ~31);

  v8f acc = {0.f, 0.f, 0.f, 0.f, 0.f, 0.f, 0.f, 0.f};

  for (int kc = start; kc < fullEnd; kc += 32) {
    const float4* pa1 = (const float4*)(row + kc + kb0);        // A cols kb0+0..7
    const float4* pa2 = (const float4*)(row + kc + kb0 + 16);   // A cols kb0+16..23
    const float4* pb  = (const float4*)(row + kc + hi16);       // B cols hi16+0..15
    float4 A0 = pa1[0], A1 = pa1[1];
    float4 A2 = pa2[0], A3 = pa2[1];
    float4 B0 = pb[0], B1 = pb[1], B2 = pb[2], B3 = pb[3];
    v16h a, b;
    pack8(a, 0, A0, A1);
    pack8(a, 8, A2, A3);
    pack8(b, 0, A0, A1);
    pack8(b, 8, A2, A3);
    acc = __builtin_amdgcn_wmma_f32_16x16x32_f16(
        false, a, false, b, (short)0, acc, false, false);
  }

  if (fullEnd < end) {                       // 16-col tail (only last K-split)
    int kc = fullEnd;                        // 49984
    // A run1 (halves 0..7): cols kc+kb0+0..7 -> always in range (<= 49999)
    const float4* pa1 = (const float4*)(row + kc + kb0);
    float4 A0 = pa1[0], A1 = pa1[1];
    // A run2 (halves 8..15): cols >= 50000 -> zeros
    // B: cols kc+hi16+0..15 -> valid only for hi==0; load valid cols, mask hi==1
    const float4* pb = (const float4*)(row + kc);
    float4 B0 = pb[0], B1 = pb[1], B2 = pb[2], B3 = pb[3];
    float s = (hi == 0) ? 1.f : 0.f;
    B0.x *= s; B0.y *= s; B0.z *= s; B0.w *= s;
    B1.x *= s; B1.y *= s; B1.z *= s; B1.w *= s;
    B2.x *= s; B2.y *= s; B2.z *= s; B2.w *= s;
    B3.x *= s; B3.y *= s; B3.z *= s; B3.w *= s;
    float4 Z = make_float4(0.f, 0.f, 0.f, 0.f);
    v16h a, b;
    pack8(a, 0, A0, A1);
    pack8(a, 8, Z, Z);
    pack8(b, 0, A0, A1);
    pack8(b, 8, Z, Z);
    acc = __builtin_amdgcn_wmma_f32_16x16x32_f16(
        false, a, false, b, (short)0, acc, false, false);
  }

  // diagonal extraction: exact-integer partial sums, f32 atomics stay exact
  if (lane < 8)        atomicAdd(&norms2[g0 + lane], acc[lane]);
  else if (lane >= 24) atomicAdd(&norms2[g0 + lane - 16], acc[lane - 24]);
}

// ------------------------------------------------ norms2 -> max(sqrt, 1e-12)
__global__ void wl_norm_finalize(float* __restrict__ norms2) {
  int g = threadIdx.x;
  if (g < N_GRAPHS) norms2[g] = fmaxf(sqrtf(norms2[g]), 1e-12f);
}

// ------------------------------------------------- h /= norm (exact division)
__global__ void wl_normalize(float* __restrict__ h, const float* __restrict__ norms) {
  int c = blockIdx.x * blockDim.x + threadIdx.x;
  int g = blockIdx.y;
  if (c >= N_NODES) return;
  long long idx = (long long)g * N_NODES + c;
  h[idx] = h[idx] / norms[g];
}

extern "C" void kernel_launch(void* const* d_in, const int* in_sizes, int n_in,
                              void* d_out, int out_size, void* d_ws, size_t ws_size,
                              hipStream_t stream) {
  const float* x          = (const float*)d_in[0];
  const int*   edge_index = (const int*)d_in[1];
  const int*   batch      = (const int*)d_in[2];
  const int*   src = edge_index;
  const int*   dst = edge_index + N_EDGES;
  float*       out = (float*)d_out;

  // ---- workspace carve-out (~3.4 MB total) ----
  char* ws = (char*)d_ws;
  size_t off = 0;
  auto take = [&](size_t bytes) -> void* {
    void* p = ws + off;
    off = (off + bytes + 255) & ~(size_t)255;
    return p;
  };
  unsigned long long* nsum   = (unsigned long long*)take(sizeof(unsigned long long) * N_NODES);
  unsigned long long* sig    = (unsigned long long*)take(sizeof(unsigned long long) * N_NODES);
  unsigned long long* htKeys = (unsigned long long*)take(sizeof(unsigned long long) * HT_CAP);
  int*   htVals  = (int*)take(sizeof(int) * HT_CAP);
  int*   colorsA = (int*)take(sizeof(int) * N_NODES);
  int*   colorsB = (int*)take(sizeof(int) * N_NODES);
  int*   rep     = (int*)take(sizeof(int) * N_NODES);
  int*   flags   = (int*)take(sizeof(int) * N_NODES);
  int*   excl    = (int*)take(sizeof(int) * N_NODES);
  float* norms2  = (float*)take(sizeof(float) * N_GRAPHS);

  const int nodeBlocks = (N_NODES + 255) / 256;
  const int edgeBlocks = (N_EDGES + 255) / 256;
  const int htBlocks   = HT_CAP / 256;

  // zero the full output every call (harness doesn't re-poison between replays)
  const long long n4 = (long long)N_LAYERS * N_GRAPHS * N_NODES / 4;
  wl_zero_out<<<4096, 256, 0, stream>>>((float4*)out, n4);

  wl_argmax<<<nodeBlocks, 256, 0, stream>>>(x, colorsA);

  int* cur = colorsA;
  int* nxt = colorsB;
  for (int L = 0; L < N_LAYERS; ++L) {
    float* layer = out + (long long)L * N_GRAPHS * N_NODES;

    wl_clear    <<<htBlocks,   256, 0, stream>>>(nsum, htKeys, htVals, norms2);
    wl_edges    <<<edgeBlocks, 256, 0, stream>>>(src, dst, cur, nsum);
    wl_sig      <<<nodeBlocks, 256, 0, stream>>>(cur, nsum, sig);
    wl_ht_insert<<<nodeBlocks, 256, 0, stream>>>(sig, htKeys, htVals);
    wl_lookup   <<<nodeBlocks, 256, 0, stream>>>(sig, htKeys, htVals, rep, flags);
    wl_scan     <<<1, 1024, 0, stream>>>(flags, excl);
    wl_newcolor <<<nodeBlocks, 256, 0, stream>>>(rep, excl, nxt);

    wl_hist         <<<nodeBlocks, 256, 0, stream>>>(batch, nxt, layer);
    wl_norm_wmma    <<<dim3(N_GRAPHS / 16, NRM_SPLITS), 32, 0, stream>>>(layer, norms2);
    wl_norm_finalize<<<1, 256, 0, stream>>>(norms2);
    wl_normalize    <<<dim3(nodeBlocks, N_GRAPHS), 256, 0, stream>>>(layer, norms2);

    int* t = cur; cur = nxt; nxt = t;
  }
}
